// CLIPToImageTransformer_29729763623588
// MI455X (gfx1250) — matrix operand order repaired and checked by reading, not compiled, not run on hardware
//
#include <hip/hip_runtime.h>
#include <hip/hip_bf16.h>
#include <math.h>

#define NL     24
#define DMODEL 1024
#define NH     16
#define DHEAD  64
#define DFF    4096
#define CLIPD  512
#define VOCAB  16384
#define SEQ    257
#define SLD    260          // padded leading dim for scores rows (16B aligned)
#define BB     8
#define LTOK   255
#define SB     (SEQ * BB)   // 2056 rows

typedef __attribute__((ext_vector_type(16))) __bf16 v16bf;
typedef __attribute__((ext_vector_type(8)))  __bf16 v8bf;
typedef __attribute__((ext_vector_type(8)))  float  v8f;
typedef __attribute__((ext_vector_type(4)))  float  f4;

struct GemmP {
  const float* A;  long long lda;
  const void*  Bm; long long ldb;   // fp32 or bf16 depending on template
  const float* bias;
  float* C;
  int M, N, K;
  float alpha;
  int Hdiv;                       // z -> (b = z/Hdiv, h = z%Hdiv)
  long long oaB, oaH, obB, obH, ocB, ocH;
  int mdiv; long long csQ, csR;   // Caddr = (m/mdiv)*csQ + (m%mdiv)*csR + n
};

__device__ __forceinline__ int imind(int a, int b) { return a < b ? a : b; }

// ---------------------------------------------------------------------------
// Batched GEMM on the WMMA path. Wave tile = 32(M) x 64(N): 2 A-frags x 4
// B-frags -> 8 v_wmma_f32_16x16x32_bf16 per K-step. A is fp32 converted to
// bf16 in registers; B is fp32 (BBF=0) or pre-converted bf16 (BBF=1, loads 16
// contiguous bf16 = 2x b128, zero conversions). Main K-loop is branch-free;
// M/N edges via address clamping (stores guarded), K tail (attn@V, K=257)
// via one masked step.
// TRANSB=1: B_elem(k,n) = B[n*ldb + k]   TRANSB=0: B_elem(k,n) = B[k*ldb + n]
// EPI=1: exact GELU after bias.
// ---------------------------------------------------------------------------
template <bool TRANSB, int EPI, bool BBF>
__global__ __launch_bounds__(256) void gemm_wmma(GemmP p) {
  const int lane = threadIdx.x & 31;
  const int wid  = threadIdx.x >> 5;
  const int z    = blockIdx.z;
  const int zb = z / p.Hdiv, zh = z % p.Hdiv;
  const float* A = p.A + (long long)zb * p.oaB + (long long)zh * p.oaH;
  const float*  Bf = (const float*)p.Bm + (BBF ? 0 : ((long long)zb * p.obB + (long long)zh * p.obH));
  const __bf16* Bh = (const __bf16*)p.Bm;
  float* C = p.C + (long long)zb * p.ocB + (long long)zh * p.ocH;

  const int tilesN = (p.N + 63) >> 6;
  const int tilesM = (p.M + 31) >> 5;
  const int t = blockIdx.x * 8 + wid;          // wave-uniform
  if (t >= tilesM * tilesN) return;            // whole-wave exit: EXEC stays full
  const int tn = t % tilesN, tm = t / tilesN;

  const int hlf = lane >> 4;                   // lane half (0|1)
  const int l16 = lane & 15;

  // Row-clamped A row pointers (M-edge garbage is never stored).
  const float* arow0 = A + (long long)imind((tm << 5) + l16,      p.M - 1) * p.lda;
  const float* arow1 = A + (long long)imind((tm << 5) + 16 + l16, p.M - 1) * p.lda;
  // Column-clamped B column indices.
  const int nc[4] = { imind((tn << 6) + l16,      p.N - 1),
                      imind((tn << 6) + 16 + l16, p.N - 1),
                      imind((tn << 6) + 32 + l16, p.N - 1),
                      imind((tn << 6) + 48 + l16, p.N - 1) };

  v8f acc[2][4] = {{v8f{}, v8f{}, v8f{}, v8f{}}, {v8f{}, v8f{}, v8f{}, v8f{}}};

  const int kmain = p.K & ~31;
  for (int kb = 0; kb < kmain; kb += 32) {
    // ---- A fragments: two contiguous 8-float runs per lane (b128 x2 each) --
    v16bf af0, af1;
    {
      const f4* q0 = (const f4*)(arow0 + kb + (hlf << 4));
      const f4* q1 = (const f4*)(arow0 + kb + 8 + (hlf << 4));
      const f4* r0 = (const f4*)(arow1 + kb + (hlf << 4));
      const f4* r1 = (const f4*)(arow1 + kb + 8 + (hlf << 4));
      f4 a0 = q0[0], a1 = q0[1], a2 = q1[0], a3 = q1[1];
      f4 b0 = r0[0], b1 = r0[1], b2 = r1[0], b3 = r1[1];
#pragma unroll
      for (int i = 0; i < 4; ++i) {
        af0[i]      = (__bf16)a0[i];  af0[i + 4]  = (__bf16)a1[i];
        af0[i + 8]  = (__bf16)a2[i];  af0[i + 12] = (__bf16)a3[i];
        af1[i]      = (__bf16)b0[i];  af1[i + 4]  = (__bf16)b1[i];
        af1[i + 8]  = (__bf16)b2[i];  af1[i + 12] = (__bf16)b3[i];
      }
    }
    if (kb + 32 < kmain) {
      __builtin_prefetch(arow0 + kb + 32, 0, 1);   // global_prefetch_b8
      __builtin_prefetch(arow1 + kb + 32, 0, 1);
    }

#pragma unroll
    for (int j = 0; j < 4; ++j) {
      // ---- B fragment: lane = N column, K = hlf*16 + i ---------------------
      v16bf bfrag;
      if (BBF) {
        // 16 contiguous bf16 (32B, aligned by construction): direct load
        bfrag = *(const v16bf*)(Bh + (long long)nc[j] * p.ldb + kb + (hlf << 4));
      } else if (TRANSB) {
        const f4* br = (const f4*)(Bf + (long long)nc[j] * p.ldb + kb + (hlf << 4));
        f4 c0 = br[0], c1 = br[1], c2 = br[2], c3 = br[3];
#pragma unroll
        for (int i = 0; i < 4; ++i) {
          bfrag[i]      = (__bf16)c0[i];  bfrag[i + 4]  = (__bf16)c1[i];
          bfrag[i + 8]  = (__bf16)c2[i];  bfrag[i + 12] = (__bf16)c3[i];
        }
      } else {
#pragma unroll
        for (int i = 0; i < 16; ++i) {
          const int kk = kb + (hlf << 4) + i;
          bfrag[i] = (__bf16)Bf[(long long)kk * p.ldb + nc[j]];
        }
      }
      acc[0][j] = __builtin_amdgcn_wmma_f32_16x16x32_bf16(
          false, af0, false, bfrag, (short)0, acc[0][j], false, false);
      acc[1][j] = __builtin_amdgcn_wmma_f32_16x16x32_bf16(
          false, af1, false, bfrag, (short)0, acc[1][j], false, false);
    }
  }

  if (kmain < p.K) {   // cold K tail: clamped addresses + value masking
    v16bf af0, af1;
#pragma unroll
    for (int i = 0; i < 16; ++i) {
      const int kk = kmain + (hlf << 4) + i;
      const int kc = imind(kk, p.K - 1);
      const bool ok = kk < p.K;
      af0[i] = (__bf16)(ok ? arow0[kc] : 0.0f);
      af1[i] = (__bf16)(ok ? arow1[kc] : 0.0f);
    }
#pragma unroll
    for (int j = 0; j < 4; ++j) {
      v16bf bfrag;
#pragma unroll
      for (int i = 0; i < 16; ++i) {
        const int kk = kmain + (hlf << 4) + i;
        const int kc = imind(kk, p.K - 1);
        float v;
        if (BBF)         v = (float)Bh[(long long)nc[j] * p.ldb + kc];
        else if (TRANSB) v = Bf[(long long)nc[j] * p.ldb + kc];
        else             v = Bf[(long long)kc * p.ldb + nc[j]];
        bfrag[i] = (__bf16)(kk < p.K ? v : 0.0f);
      }
      acc[0][j] = __builtin_amdgcn_wmma_f32_16x16x32_bf16(
          false, af0, false, bfrag, (short)0, acc[0][j], false, false);
      acc[1][j] = __builtin_amdgcn_wmma_f32_16x16x32_bf16(
          false, af1, false, bfrag, (short)0, acc[1][j], false, false);
    }
  }

  // ---- epilogue: alpha, bias, optional GELU, permuted guarded store --------
#pragma unroll
  for (int j = 0; j < 4; ++j) {
    const int n = (tn << 6) + (j << 4) + l16;
    if (n >= p.N) continue;
    const float bv = p.bias ? p.bias[n] : 0.0f;
#pragma unroll
    for (int u = 0; u < 2; ++u) {
#pragma unroll
      for (int r = 0; r < 8; ++r) {
        const int m = (tm << 5) + (u << 4) + (hlf << 3) + r;  // M = r + 8*half
        if (m >= p.M) continue;
        float v = p.alpha * acc[u][j][r] + bv;
        if (EPI == 1) v = 0.5f * v * (1.0f + erff(v * 0.70710678118654752f));
        const long long addr = (long long)(m / p.mdiv) * p.csQ +
                               (long long)(m % p.mdiv) * p.csR + n;
        C[addr] = v;
      }
    }
  }
}

// ---------------------------------------------------------------------------
// fp32 -> bf16 weight conversion (8 elems/thread, b128 in / b128 out)
// ---------------------------------------------------------------------------
__global__ __launch_bounds__(256) void cvt_bf16_kernel(
    const float* __restrict__ src, __bf16* __restrict__ dst, long long n) {
  const long long i = ((long long)blockIdx.x * 256 + threadIdx.x) * 8;
  if (i >= n) return;
  f4 a = *(const f4*)(src + i);
  f4 b = *(const f4*)(src + i + 4);
  v8bf o;
#pragma unroll
  for (int k = 0; k < 4; ++k) { o[k] = (__bf16)a[k]; o[k + 4] = (__bf16)b[k]; }
  *(v8bf*)(dst + i) = o;
}

// ---------------------------------------------------------------------------
// Embedding: x[s,b,:] = {ce_proj | cs_proj | tok_embed} + pos_emb[s]
// ---------------------------------------------------------------------------
__global__ __launch_bounds__(256) void embed_kernel(
    const float* __restrict__ clip_embed, const float* __restrict__ clip_score,
    const int* __restrict__ toks, const float* __restrict__ clip_in_w,
    const float* __restrict__ score_in_w, const float* __restrict__ in_embed,
    const float* __restrict__ pos_emb, float* __restrict__ x) {
  const int sb = blockIdx.x;            // s*BB + b
  const int s = sb / BB, b = sb % BB;
  __shared__ float red[256];
  float scale = 0.0f;
  if (s == 0) {
    float ss = 0.0f;
    for (int c = threadIdx.x; c < CLIPD; c += 256) {
      const float v = clip_embed[b * CLIPD + c];
      ss += v * v;
    }
    red[threadIdx.x] = ss; __syncthreads();
    for (int o = 128; o > 0; o >>= 1) {
      if (threadIdx.x < o) red[threadIdx.x] += red[threadIdx.x + o];
      __syncthreads();
    }
    const float nrm = fmaxf(sqrtf(red[0]), 1e-12f);
    scale = 22.62741699796952f / nrm;   // sqrt(512)/||ce||
  }
  for (int d = threadIdx.x; d < DMODEL; d += 256) {
    float v;
    if (s == 0) {
      float acc = 0.0f;
      for (int c = 0; c < CLIPD; ++c)
        acc += clip_embed[b * CLIPD + c] * clip_in_w[(long long)d * CLIPD + c];
      v = acc * scale;
    } else if (s == 1) {
      v = clip_score[b] * score_in_w[d];
    } else {
      const int tk = toks[b * LTOK + (s - 2)];
      v = in_embed[(long long)tk * DMODEL + d];
    }
    x[(long long)sb * DMODEL + d] = v + pos_emb[(long long)s * DMODEL + d];
  }
}

// ---------------------------------------------------------------------------
// Masked row softmax over scores[z, s, :] (row stride SLD). Causal mask with
// the reference's mask[0,1]=0 quirk.
// ---------------------------------------------------------------------------
__global__ __launch_bounds__(256) void softmax_kernel(float* __restrict__ sc) {
  const int s = blockIdx.x;
  const long long z = blockIdx.y;
  float* row = sc + (z * SEQ + s) * (long long)SLD;
  __shared__ float red[256];
  float vals[2];
  float mx = -3.0e38f;
#pragma unroll
  for (int i = 0; i < 2; ++i) {
    const int tt = threadIdx.x + i * 256;
    const bool ok = (tt < SEQ) && ((tt <= s) || (s == 0 && tt == 1));
    vals[i] = ok ? row[tt] : -3.0e38f;
    if (ok) mx = fmaxf(mx, vals[i]);
  }
  red[threadIdx.x] = mx; __syncthreads();
  for (int o = 128; o > 0; o >>= 1) {
    if (threadIdx.x < o) red[threadIdx.x] = fmaxf(red[threadIdx.x], red[threadIdx.x + o]);
    __syncthreads();
  }
  mx = red[0]; __syncthreads();
  float sum = 0.0f;
#pragma unroll
  for (int i = 0; i < 2; ++i) {
    const int tt = threadIdx.x + i * 256;
    const bool ok = (tt < SEQ) && ((tt <= s) || (s == 0 && tt == 1));
    vals[i] = ok ? __expf(vals[i] - mx) : 0.0f;
    sum += vals[i];
  }
  red[threadIdx.x] = sum; __syncthreads();
  for (int o = 128; o > 0; o >>= 1) {
    if (threadIdx.x < o) red[threadIdx.x] += red[threadIdx.x + o];
    __syncthreads();
  }
  const float inv = 1.0f / red[0];
#pragma unroll
  for (int i = 0; i < 2; ++i) {
    const int tt = threadIdx.x + i * 256;
    if (tt < SEQ) row[tt] = vals[i] * inv;
  }
}

// ---------------------------------------------------------------------------
// out[row] = LayerNorm(xin[row] + add[row]) * g + b     (one block per row)
// ---------------------------------------------------------------------------
__global__ __launch_bounds__(256) void resln_kernel(
    const float* __restrict__ xin, const float* __restrict__ addv,
    const float* __restrict__ g, const float* __restrict__ bta,
    float* __restrict__ xout) {
  const long long row = blockIdx.x;
  const float* xr = xin + row * DMODEL;
  const float* ar = addv + row * DMODEL;
  float y[4];
  float sum = 0.0f;
#pragma unroll
  for (int i = 0; i < 4; ++i) {
    const int d = threadIdx.x + i * 256;
    y[i] = xr[d] + ar[d];
    sum += y[i];
  }
  __shared__ float red[256];
  red[threadIdx.x] = sum; __syncthreads();
  for (int o = 128; o > 0; o >>= 1) {
    if (threadIdx.x < o) red[threadIdx.x] += red[threadIdx.x + o];
    __syncthreads();
  }
  const float mean = red[0] * (1.0f / DMODEL); __syncthreads();
  float vs = 0.0f;
#pragma unroll
  for (int i = 0; i < 4; ++i) { const float dd = y[i] - mean; vs += dd * dd; }
  red[threadIdx.x] = vs; __syncthreads();
  for (int o = 128; o > 0; o >>= 1) {
    if (threadIdx.x < o) red[threadIdx.x] += red[threadIdx.x + o];
    __syncthreads();
  }
  const float inv = rsqrtf(red[0] * (1.0f / DMODEL) + 1e-5f);
#pragma unroll
  for (int i = 0; i < 4; ++i) {
    const int d = threadIdx.x + i * 256;
    xout[row * DMODEL + d] = (y[i] - mean) * inv * g[d] + bta[d];
  }
}

// ---------------------------------------------------------------------------
static inline void launch_gemm(bool transb, int epi, bool bbf, const GemmP& p,
                               int zdim, hipStream_t st) {
  const int tiles = ((p.M + 31) / 32) * ((p.N + 63) / 64);
  dim3 grid((tiles + 7) / 8, 1, zdim), blk(256);
  if (bbf) {
    if (epi == 1) gemm_wmma<true, 1, true><<<grid, blk, 0, st>>>(p);
    else          gemm_wmma<true, 0, true><<<grid, blk, 0, st>>>(p);
  } else if (transb) {
    if (epi == 1) gemm_wmma<true, 1, false><<<grid, blk, 0, st>>>(p);
    else          gemm_wmma<true, 0, false><<<grid, blk, 0, st>>>(p);
  } else {
    gemm_wmma<false, 0, false><<<grid, blk, 0, st>>>(p);
  }
}

static inline void launch_cvt(const float* s, __bf16* d, long long n,
                              hipStream_t st) {
  const long long blocks = (n / 8 + 255) / 256;
  cvt_bf16_kernel<<<(unsigned)blocks, 256, 0, st>>>(s, d, n);
}

extern "C" void kernel_launch(void* const* d_in, const int* in_sizes, int n_in,
                              void* d_out, int out_size, void* d_ws, size_t ws_size,
                              hipStream_t stream) {
  const float* clip_embed = (const float*)d_in[0];
  const float* clip_score = (const float*)d_in[1];
  const int*   toks       = (const int*)d_in[2];
  const float* clip_in_w  = (const float*)d_in[3];
  const float* score_in_w = (const float*)d_in[4];
  const float* in_embed   = (const float*)d_in[5];
  const float* pos_emb    = (const float*)d_in[6];
  const float* Wqkv = (const float*)d_in[7];
  const float* bqkv = (const float*)d_in[8];
  const float* Wo   = (const float*)d_in[9];
  const float* bo   = (const float*)d_in[10];
  const float* W1   = (const float*)d_in[11];
  const float* b1   = (const float*)d_in[12];
  const float* W2   = (const float*)d_in[13];
  const float* b2   = (const float*)d_in[14];
  const float* ln1g = (const float*)d_in[15];
  const float* ln1b = (const float*)d_in[16];
  const float* ln2g = (const float*)d_in[17];
  const float* ln2b = (const float*)d_in[18];
  const float* out_w= (const float*)d_in[19];
  const float* out_b= (const float*)d_in[20];

  float* logits = (float*)d_out;                                  // [B, S-1, V]
  float* cache  = logits + (size_t)BB * (SEQ - 1) * VOCAB;        // [24, S, B, D]

  float* ws = (float*)d_ws;
  size_t off = 0;
  float* x0     = ws + off; off += (size_t)SB * DMODEL;
  float* qkv    = ws + off; off += (size_t)SB * 3 * DMODEL;
  float* scores = ws + off; off += (size_t)BB * NH * SEQ * SLD + 1024;  // padded
  float* ctx    = ws + off; off += (size_t)SB * DMODEL;
  float* abuf   = ws + off; off += (size_t)SB * DMODEL;
  float* xmid   = ws + off; off += (size_t)SB * DMODEL;
  float* ffh    = ws + off; off += (size_t)SB * DFF;
  float* ffo    = ws + off; off += (size_t)SB * DMODEL;
  off = (off + 63) & ~(size_t)63;                 // 32B-align bf16 region

  // Optional pre-converted bf16 weights (halves B-side traffic, no hot cvts).
  const size_t nWqkv = (size_t)NL * 3 * DMODEL * DMODEL;
  const size_t nWo   = (size_t)NL * DMODEL * DMODEL;
  const size_t nW1   = (size_t)NL * DFF * DMODEL;
  const size_t nW2   = (size_t)NL * DMODEL * DFF;
  const size_t nOw   = (size_t)VOCAB * DMODEL;
  const size_t nbf   = nWqkv + nWo + nW1 + nW2 + nOw;
  const size_t need  = off * sizeof(float) + nbf * sizeof(__bf16);
  const bool   bbf   = ws_size >= need;

  __bf16* wbase = (__bf16*)(ws + off);
  __bf16* qkv_bf = wbase;
  __bf16* wo_bf  = qkv_bf + nWqkv;
  __bf16* w1_bf  = wo_bf  + nWo;
  __bf16* w2_bf  = w1_bf  + nW1;
  __bf16* ow_bf  = w2_bf  + nW2;
  (void)in_sizes; (void)n_in; (void)out_size;

  if (bbf) {
    launch_cvt(Wqkv, qkv_bf, (long long)nWqkv, stream);
    launch_cvt(Wo,   wo_bf,  (long long)nWo,   stream);
    launch_cvt(W1,   w1_bf,  (long long)nW1,   stream);
    launch_cvt(W2,   w2_bf,  (long long)nW2,   stream);
    launch_cvt(out_w,ow_bf,  (long long)nOw,   stream);
  }

  embed_kernel<<<SB, 256, 0, stream>>>(clip_embed, clip_score, toks, clip_in_w,
                                       score_in_w, in_embed, pos_emb, x0);

  for (int l = 0; l < NL; ++l) {
    const float* xin = (l == 0) ? x0 : (cache + (size_t)(l - 1) * SB * DMODEL);

    { // qkv = xin @ Wqkv[l]^T + bqkv[l]            [SB, 3D]
      GemmP p{}; p.A = xin; p.lda = DMODEL;
      p.Bm = bbf ? (const void*)(qkv_bf + (size_t)l * 3 * DMODEL * DMODEL)
                 : (const void*)(Wqkv   + (size_t)l * 3 * DMODEL * DMODEL);
      p.ldb = DMODEL;
      p.bias = bqkv + (size_t)l * 3 * DMODEL; p.C = qkv;
      p.M = SB; p.N = 3 * DMODEL; p.K = DMODEL; p.alpha = 1.0f;
      p.Hdiv = 1; p.mdiv = 1; p.csQ = 3 * DMODEL; p.csR = 0;
      launch_gemm(true, 0, bbf, p, 1, stream);
    }
    { // scores[z] = Q[z] @ K[z]^T / 8             z = b*NH+h, [S, S] (ld SLD)
      GemmP p{}; p.A = qkv; p.lda = (long long)BB * 3 * DMODEL;
      p.Bm = qkv + DMODEL; p.ldb = (long long)BB * 3 * DMODEL;
      p.bias = nullptr; p.C = scores;
      p.M = SEQ; p.N = SEQ; p.K = DHEAD; p.alpha = 0.125f;
      p.Hdiv = NH;
      p.oaB = 3 * DMODEL; p.oaH = DHEAD; p.obB = 3 * DMODEL; p.obH = DHEAD;
      p.ocB = (long long)NH * SEQ * SLD; p.ocH = (long long)SEQ * SLD;
      p.mdiv = 1; p.csQ = SLD; p.csR = 0;
      launch_gemm(true, 0, false, p, BB * NH, stream);
    }
    softmax_kernel<<<dim3(SEQ, BB * NH), 256, 0, stream>>>(scores);
    { // ctx[z] = P[z] @ V[z]                      [S, 64] -> ctx[S,B,D]
      GemmP p{}; p.A = scores; p.lda = SLD;
      p.Bm = qkv + 2 * DMODEL; p.ldb = (long long)BB * 3 * DMODEL;
      p.bias = nullptr; p.C = ctx;
      p.M = SEQ; p.N = DHEAD; p.K = SEQ; p.alpha = 1.0f;
      p.Hdiv = NH;
      p.oaB = (long long)NH * SEQ * SLD; p.oaH = (long long)SEQ * SLD;
      p.obB = 3 * DMODEL; p.obH = DHEAD;
      p.ocB = DMODEL; p.ocH = DHEAD;
      p.mdiv = 1; p.csQ = (long long)BB * DMODEL; p.csR = 0;
      launch_gemm(false, 0, false, p, BB * NH, stream);
    }
    { // a = ctx @ Wo[l]^T + bo[l]                 [SB, D]
      GemmP p{}; p.A = ctx; p.lda = DMODEL;
      p.Bm = bbf ? (const void*)(wo_bf + (size_t)l * DMODEL * DMODEL)
                 : (const void*)(Wo    + (size_t)l * DMODEL * DMODEL);
      p.ldb = DMODEL;
      p.bias = bo + (size_t)l * DMODEL; p.C = abuf;
      p.M = SB; p.N = DMODEL; p.K = DMODEL; p.alpha = 1.0f;
      p.Hdiv = 1; p.mdiv = 1; p.csQ = DMODEL; p.csR = 0;
      launch_gemm(true, 0, bbf, p, 1, stream);
    }
    resln_kernel<<<SB, 256, 0, stream>>>(xin, abuf, ln1g + (size_t)l * DMODEL,
                                         ln1b + (size_t)l * DMODEL, xmid);
    { // h = GELU(xmid @ W1[l]^T + b1[l])          [SB, FF]
      GemmP p{}; p.A = xmid; p.lda = DMODEL;
      p.Bm = bbf ? (const void*)(w1_bf + (size_t)l * DFF * DMODEL)
                 : (const void*)(W1    + (size_t)l * DFF * DMODEL);
      p.ldb = DMODEL;
      p.bias = b1 + (size_t)l * DFF; p.C = ffh;
      p.M = SB; p.N = DFF; p.K = DMODEL; p.alpha = 1.0f;
      p.Hdiv = 1; p.mdiv = 1; p.csQ = DFF; p.csR = 0;
      launch_gemm(true, 1, bbf, p, 1, stream);
    }
    { // ff = h @ W2[l]^T + b2[l]                  [SB, D]
      GemmP p{}; p.A = ffh; p.lda = DFF;
      p.Bm = bbf ? (const void*)(w2_bf + (size_t)l * DMODEL * DFF)
                 : (const void*)(W2    + (size_t)l * DMODEL * DFF);
      p.ldb = DFF;
      p.bias = b2 + (size_t)l * DMODEL; p.C = ffo;
      p.M = SB; p.N = DMODEL; p.K = DFF; p.alpha = 1.0f;
      p.Hdiv = 1; p.mdiv = 1; p.csQ = DMODEL; p.csR = 0;
      launch_gemm(true, 0, bbf, p, 1, stream);
    }
    resln_kernel<<<SB, 256, 0, stream>>>(xmid, ffo, ln2g + (size_t)l * DMODEL,
                                         ln2b + (size_t)l * DMODEL,
                                         cache + (size_t)l * SB * DMODEL);
  }

  { // logits[b, s-1, :] = x_final[s,b,:] @ out_w^T + out_b   (s >= 1)
    GemmP p{};
    p.A = cache + (size_t)(NL - 1) * SB * DMODEL + (size_t)BB * DMODEL;
    p.lda = DMODEL;
    p.Bm = bbf ? (const void*)ow_bf : (const void*)out_w;
    p.ldb = DMODEL; p.bias = out_b; p.C = logits;
    p.M = (SEQ - 1) * BB; p.N = VOCAB; p.K = DMODEL; p.alpha = 1.0f;
    p.Hdiv = 1;
    p.mdiv = BB; p.csQ = VOCAB; p.csR = (long long)(SEQ - 1) * VOCAB;
    launch_gemm(true, 0, bbf, p, 1, stream);
  }
}
